// MPNNLayer_84335977824816
// MI455X (gfx1250) — matrix operand order repaired and checked by reading, not compiled, not run on hardware
//
#include <hip/hip_runtime.h>
#include <hip/hip_bf16.h>

// ---------------------------------------------------------------------------
// MPNN layer for gfx1250 (MI455X): bf16 WMMA (v_wmma_f32_16x16x32_bf16)
// edge-MLP GEMMs + fp32 global-atomic scatter + WMMA GRU gates.
// ---------------------------------------------------------------------------

#define HID 256     // H
#define EDF 64      // ED
#define DIN 576     // 2H + ED
#define TM  64      // edges / nodes per block tile
#define AS  584     // LDS stride for 576-wide A tile (bf16 elems, 16B multiple)
#define HS  264     // LDS stride for 256-wide tiles  (bf16 elems, 16B multiple)

typedef __bf16 v16bf __attribute__((ext_vector_type(16)));
typedef float  v8f   __attribute__((ext_vector_type(8)));

union Frag16 { uint4 u[2]; v16bf v; };

__device__ __forceinline__ unsigned short bf16raw(float f) {
  unsigned int u = __float_as_uint(f);
  u += 0x7fffu + ((u >> 16) & 1u);          // round-to-nearest-even
  return (unsigned short)(u >> 16);
}
__device__ __forceinline__ unsigned int pack_bf2(float a, float b) {
  return (unsigned int)bf16raw(a) | ((unsigned int)bf16raw(b) << 16);
}

// ---------------------------------------------------------------- prep ------
__global__ void cvt_bf16_kernel(const float* __restrict__ in,
                                unsigned short* __restrict__ out, int n) {
  int i = blockIdx.x * blockDim.x + threadIdx.x;
  if (i < n) out[i] = bf16raw(in[i]);
}

__global__ void zero_f32_kernel(float* __restrict__ p, int n) {
  int i = blockIdx.x * blockDim.x + threadIdx.x;
  if (i < n) p[i] = 0.0f;
}

// ------------------------------------------------- edge MLP + scatter ------
__global__ __launch_bounds__(256)
void edge_mlp_kernel(const unsigned short* __restrict__ hbf,
                     const int* __restrict__ eidx,          // [2,E]
                     const float* __restrict__ eattr,       // [E,64]
                     const unsigned short* __restrict__ W1bf, // [256,576]
                     const float* __restrict__ b1,
                     const unsigned short* __restrict__ W2bf, // [256,256]
                     const float* __restrict__ b2,
                     float* __restrict__ agg, int E) {
  __shared__ unsigned short Atile[TM * AS];   // edge_in tile, bf16
  __shared__ unsigned short Hid[TM * HS];     // relu(layer1) tile, bf16
  __shared__ int rowsLDS[TM];

  const int tid = threadIdx.x;
  const int e0  = blockIdx.x * TM;

  if (tid < TM) rowsLDS[tid] = eidx[e0 + tid];

  // Build A tile: 64 edges x 576 bf16, 16-byte chunks (72 per edge).
  for (int i = tid; i < TM * 72; i += 256) {
    int e = i / 72, j = i % 72;
    uint4 val;
    if (j < 32) {                                   // h[row]
      int r = eidx[e0 + e];
      val = *(const uint4*)(hbf + (size_t)r * HID + j * 8);
    } else if (j < 64) {                            // h[col]
      int c = eidx[E + e0 + e];
      val = *(const uint4*)(hbf + (size_t)c * HID + (j - 32) * 8);
    } else {                                        // edge_attr (fp32 -> bf16)
      const float* s = eattr + (size_t)(e0 + e) * EDF + (j - 64) * 8;
      float4 f0 = *(const float4*)(s);
      float4 f1 = *(const float4*)(s + 4);
      val.x = pack_bf2(f0.x, f0.y); val.y = pack_bf2(f0.z, f0.w);
      val.z = pack_bf2(f1.x, f1.y); val.w = pack_bf2(f1.z, f1.w);
    }
    *(uint4*)(&Atile[e * AS + j * 8]) = val;
  }
  __syncthreads();

  const int wave  = tid >> 5;
  const int lane  = tid & 31;
  const int lm    = lane & 15;
  const int ahalf = (lane < 16) ? 0 : 16;   // A-fragment K half
  const int bhalf = (lane < 16) ? 0 : 16;   // B-fragment K half

  // ---- GEMM1: Hid = relu(A @ W1^T + b1), 64x256, K=576 ----
  for (int t = 0; t < 8; ++t) {
    int tile = wave * 8 + t;                // 64 tiles of 16x16
    int mt = tile >> 4, nt = tile & 15;
    int m = mt * 16 + lm;
    int n = nt * 16 + lm;
    v8f acc = {};
#pragma unroll
    for (int kb = 0; kb < DIN; kb += 32) {
      Frag16 a, b;
      a.u[0] = *(const uint4*)&Atile[m * AS + kb + ahalf];
      a.u[1] = *(const uint4*)&Atile[m * AS + kb + ahalf + 8];
      const unsigned short* bp = W1bf + (size_t)n * DIN + kb + bhalf;
      b.u[0] = *(const uint4*)(bp);
      b.u[1] = *(const uint4*)(bp + 8);
      acc = __builtin_amdgcn_wmma_f32_16x16x32_bf16(
          false, a.v, false, b.v, (short)0, acc, false, false);
    }
    float bias = b1[n];
#pragma unroll
    for (int v = 0; v < 8; ++v) {
      int mr = mt * 16 + v + ((lane < 16) ? 0 : 8);
      float x = acc[v] + bias;
      Hid[mr * HS + n] = bf16raw(x > 0.0f ? x : 0.0f);
    }
  }
  __syncthreads();

  // ---- GEMM2: msg = Hid @ W2^T + b2, then scatter-add by row index ----
  for (int t = 0; t < 8; ++t) {
    int tile = wave * 8 + t;
    int mt = tile >> 4, nt = tile & 15;
    int m = mt * 16 + lm;
    int n = nt * 16 + lm;
    v8f acc = {};
#pragma unroll
    for (int kb = 0; kb < HID; kb += 32) {
      Frag16 a, b;
      a.u[0] = *(const uint4*)&Hid[m * HS + kb + ahalf];
      a.u[1] = *(const uint4*)&Hid[m * HS + kb + ahalf + 8];
      const unsigned short* bp = W2bf + (size_t)n * HID + kb + bhalf;
      b.u[0] = *(const uint4*)(bp);
      b.u[1] = *(const uint4*)(bp + 8);
      acc = __builtin_amdgcn_wmma_f32_16x16x32_bf16(
          false, a.v, false, b.v, (short)0, acc, false, false);
    }
    float bias = b2[n];
#pragma unroll
    for (int v = 0; v < 8; ++v) {
      int mr  = mt * 16 + v + ((lane < 16) ? 0 : 8);
      int dst = rowsLDS[mr];
      __hip_atomic_fetch_add(&agg[(size_t)dst * HID + n], acc[v] + bias,
                             __ATOMIC_RELAXED, __HIP_MEMORY_SCOPE_AGENT);
    }
  }
}

// --------------------------------------------------------------- GRU -------
__global__ __launch_bounds__(256)
void gru_kernel(const float* __restrict__ agg,
                const float* __restrict__ h,
                const unsigned short* __restrict__ Wihbf, // [768,256]
                const float* __restrict__ bih,
                const unsigned short* __restrict__ Whhbf, // [768,256]
                const float* __restrict__ bhh,
                float* __restrict__ out, int N) {
  __shared__ unsigned short Abf[TM * HS];   // agg tile (bf16)
  __shared__ unsigned short Hbf[TM * HS];   // h tile (bf16)

  const int tid = threadIdx.x;
  const int n0  = blockIdx.x * TM;

  // Stage fp32 -> bf16 tiles (guard tail block).
  for (int i = tid; i < TM * 32; i += 256) {
    int r = i >> 5, j = i & 31;
    int node = n0 + r;
    uint4 av = {0, 0, 0, 0}, hv = {0, 0, 0, 0};
    if (node < N) {
      const float* ap = agg + (size_t)node * HID + j * 8;
      const float* hp = h   + (size_t)node * HID + j * 8;
      float4 a0 = *(const float4*)(ap), a1 = *(const float4*)(ap + 4);
      float4 h0 = *(const float4*)(hp), h1 = *(const float4*)(hp + 4);
      av.x = pack_bf2(a0.x, a0.y); av.y = pack_bf2(a0.z, a0.w);
      av.z = pack_bf2(a1.x, a1.y); av.w = pack_bf2(a1.z, a1.w);
      hv.x = pack_bf2(h0.x, h0.y); hv.y = pack_bf2(h0.z, h0.w);
      hv.z = pack_bf2(h1.x, h1.y); hv.w = pack_bf2(h1.z, h1.w);
    }
    *(uint4*)(&Abf[r * HS + j * 8]) = av;
    *(uint4*)(&Hbf[r * HS + j * 8]) = hv;
  }
  __syncthreads();

  const int wave  = tid >> 5;
  const int lane  = tid & 31;
  const int lm    = lane & 15;
  const int ahalf = (lane < 16) ? 0 : 16;
  const int bhalf = (lane < 16) ? 0 : 16;

  for (int t = 0; t < 8; ++t) {
    int unit = wave * 8 + t;                // 64 units of 16x16 output
    int mt = unit >> 4, nt = unit & 15;
    int m = mt * 16 + lm;
    int n = nt * 16 + lm;                   // column within 0..255

    v8f xr = {}, xz = {}, xn = {}, hr = {}, hz = {}, hn = {};
#pragma unroll
    for (int kb = 0; kb < HID; kb += 32) {
      Frag16 aA, aH, br, bz, bn;
      aA.u[0] = *(const uint4*)&Abf[m * HS + kb + ahalf];
      aA.u[1] = *(const uint4*)&Abf[m * HS + kb + ahalf + 8];
      aH.u[0] = *(const uint4*)&Hbf[m * HS + kb + ahalf];
      aH.u[1] = *(const uint4*)&Hbf[m * HS + kb + ahalf + 8];
      // W_ih rows: r -> n, z -> n+256, n-gate -> n+512
      const unsigned short* p;
      p = Wihbf + (size_t)(n)       * HID + kb + bhalf;
      br.u[0] = *(const uint4*)p; br.u[1] = *(const uint4*)(p + 8);
      xr = __builtin_amdgcn_wmma_f32_16x16x32_bf16(false, aA.v, false, br.v, (short)0, xr, false, false);
      p = Wihbf + (size_t)(n + 256) * HID + kb + bhalf;
      bz.u[0] = *(const uint4*)p; bz.u[1] = *(const uint4*)(p + 8);
      xz = __builtin_amdgcn_wmma_f32_16x16x32_bf16(false, aA.v, false, bz.v, (short)0, xz, false, false);
      p = Wihbf + (size_t)(n + 512) * HID + kb + bhalf;
      bn.u[0] = *(const uint4*)p; bn.u[1] = *(const uint4*)(p + 8);
      xn = __builtin_amdgcn_wmma_f32_16x16x32_bf16(false, aA.v, false, bn.v, (short)0, xn, false, false);
      p = Whhbf + (size_t)(n)       * HID + kb + bhalf;
      br.u[0] = *(const uint4*)p; br.u[1] = *(const uint4*)(p + 8);
      hr = __builtin_amdgcn_wmma_f32_16x16x32_bf16(false, aH.v, false, br.v, (short)0, hr, false, false);
      p = Whhbf + (size_t)(n + 256) * HID + kb + bhalf;
      bz.u[0] = *(const uint4*)p; bz.u[1] = *(const uint4*)(p + 8);
      hz = __builtin_amdgcn_wmma_f32_16x16x32_bf16(false, aH.v, false, bz.v, (short)0, hz, false, false);
      p = Whhbf + (size_t)(n + 512) * HID + kb + bhalf;
      bn.u[0] = *(const uint4*)p; bn.u[1] = *(const uint4*)(p + 8);
      hn = __builtin_amdgcn_wmma_f32_16x16x32_bf16(false, aH.v, false, bn.v, (short)0, hn, false, false);
    }

    float bxr = bih[n], bxz = bih[n + 256], bxn = bih[n + 512];
    float bhr = bhh[n], bhz = bhh[n + 256], bhn = bhh[n + 512];
#pragma unroll
    for (int v = 0; v < 8; ++v) {
      int node = n0 + mt * 16 + v + ((lane < 16) ? 0 : 8);
      if (node < N) {
        float r  = 1.0f / (1.0f + __expf(-((xr[v] + bxr) + (hr[v] + bhr))));
        float z  = 1.0f / (1.0f + __expf(-((xz[v] + bxz) + (hz[v] + bhz))));
        float nn = tanhf((xn[v] + bxn) + r * (hn[v] + bhn));
        float hv = h[(size_t)node * HID + n];
        out[(size_t)node * HID + n] = (1.0f - z) * nn + z * hv;
      }
    }
  }
}

// ------------------------------------------------------------- launch ------
static inline size_t align256(size_t x) { return (x + 255) & ~(size_t)255; }

extern "C" void kernel_launch(void* const* d_in, const int* in_sizes, int n_in,
                              void* d_out, int out_size, void* d_ws, size_t ws_size,
                              hipStream_t stream) {
  const float* h     = (const float*)d_in[0];
  const int*   eidx  = (const int*)d_in[1];
  const float* eattr = (const float*)d_in[2];
  const float* W1    = (const float*)d_in[3];
  const float* b1    = (const float*)d_in[4];
  const float* W2    = (const float*)d_in[5];
  const float* b2    = (const float*)d_in[6];
  const float* Wih   = (const float*)d_in[7];
  const float* bih   = (const float*)d_in[8];
  const float* Whh   = (const float*)d_in[9];
  const float* bhh   = (const float*)d_in[10];

  const int N = in_sizes[0] / HID;
  const int E = in_sizes[1] / 2;
  float* out = (float*)d_out;

  // Workspace layout
  char* ws = (char*)d_ws;
  size_t off = 0;
  float* agg = (float*)(ws + off);           off += align256((size_t)N * HID * 4);
  unsigned short* hbf   = (unsigned short*)(ws + off); off += align256((size_t)N * HID * 2);
  unsigned short* W1bf  = (unsigned short*)(ws + off); off += align256((size_t)HID * DIN * 2);
  unsigned short* W2bf  = (unsigned short*)(ws + off); off += align256((size_t)HID * HID * 2);
  unsigned short* Wihbf = (unsigned short*)(ws + off); off += align256((size_t)3 * HID * HID * 2);
  unsigned short* Whhbf = (unsigned short*)(ws + off); off += align256((size_t)3 * HID * HID * 2);

  auto blocks = [](int n) { return (n + 255) / 256; };

  zero_f32_kernel<<<blocks(N * HID), 256, 0, stream>>>(agg, N * HID);
  cvt_bf16_kernel<<<blocks(N * HID), 256, 0, stream>>>(h, hbf, N * HID);
  cvt_bf16_kernel<<<blocks(HID * DIN), 256, 0, stream>>>(W1, W1bf, HID * DIN);
  cvt_bf16_kernel<<<blocks(HID * HID), 256, 0, stream>>>(W2, W2bf, HID * HID);
  cvt_bf16_kernel<<<blocks(3 * HID * HID), 256, 0, stream>>>(Wih, Wihbf, 3 * HID * HID);
  cvt_bf16_kernel<<<blocks(3 * HID * HID), 256, 0, stream>>>(Whh, Whhbf, 3 * HID * HID);

  edge_mlp_kernel<<<E / TM, 256, 0, stream>>>(hbf, eidx, eattr, W1bf, b1, W2bf, b2, agg, E);
  gru_kernel<<<(N + TM - 1) / TM, 256, 0, stream>>>(agg, h, Wihbf, bih, Whhbf, bhh, out, N);
}
